// MultiHeadSelfAttention_15401752724124
// MI455X (gfx1250) — matrix operand order repaired and checked by reading, not compiled, not run on hardware
//
#include <hip/hip_runtime.h>

// MI455X / gfx1250 MHSA: v_wmma_f32_16x16x32_bf16 everywhere, bf16 operands
// staged once in workspace; GEMM B-tiles double-buffered into LDS with
// GLOBAL_LOAD_ASYNC_TO_LDS_B128 (ASYNCcnt); flash-style causal attention.

#define D_MODEL 1024
#define SEQ     2048
#define NHEAD   16
#define DK      64
#define KSTAGE  64

typedef __attribute__((ext_vector_type(16))) __bf16 v16bf;
typedef __attribute__((ext_vector_type(8)))  float  v8f;
typedef __attribute__((ext_vector_type(4)))  unsigned u32x4;

// ---- bf16 helpers (hardware packed convert, RNE) ------------------------
__device__ __forceinline__ unsigned pack_bf16x2(float lo, float hi) {
  unsigned d;
  asm("v_cvt_pk_bf16_f32 %0, %1, %2" : "=v"(d) : "v"(lo), "v"(hi));
  return d;
}

__device__ __forceinline__ __bf16 f2bf(float f) {
  unsigned d;
  asm("v_cvt_pk_bf16_f32 %0, %1, 0" : "=v"(d) : "v"(f));
  union { unsigned u; __bf16 b[2]; } o;
  o.u = d;
  return o.b[0];
}

// Load a 16-element bf16 fragment as two 16-byte vector loads.
__device__ __forceinline__ v16bf ldfrag(const __bf16* p0, const __bf16* p1) {
  union { v16bf v; u32x4 q[2]; } f;
  f.q[0] = *(const u32x4*)p0;
  f.q[1] = *(const u32x4*)p1;
  return f.v;
}

__device__ __forceinline__ v8f wmma_bf16(v16bf a, v16bf b, v8f c) {
  return __builtin_amdgcn_wmma_f32_16x16x32_bf16(false, a, false, b, (short)0, c,
                                                 false, false);
}

// Async DMA: global -> LDS, 16 bytes per lane, tracked on ASYNCcnt.
__device__ __forceinline__ void async_b128(unsigned lds_off, const void* gptr) {
  asm volatile("global_load_async_to_lds_b128 %0, %1, off"
               :: "v"(lds_off), "v"((unsigned long long)(uintptr_t)gptr)
               : "memory");
}

// ---- one-time f32 -> bf16 conversion ------------------------------------
__global__ __launch_bounds__(256) void cvt_bf16_kernel(const float* __restrict__ src,
                                                       __bf16* __restrict__ dst,
                                                       int npairs) {
  const int i = blockIdx.x * blockDim.x + threadIdx.x;
  if (i >= npairs) return;
  const float2 v = ((const float2*)src)[i];
  ((unsigned*)dst)[i] = pack_bf16x2(v.x, v.y);
}

// ---- GEMM: OUT = A (bf16, MxK) @ B^T (bf16, NxK row-major) --------------
// Block: 8 waves x (32M x 64N) = 256M x 64N. B tile (64N x 64K) is shared by
// all waves -> double-buffered async DMA into LDS, overlapped with WMMAs.
// MODE 0: store f32 [M][N]; MODE 1: store bf16 [M][N];
// MODE 2: store bf16 V^T per head: [h][dv][seq].
template <int MODE>
__global__ __launch_bounds__(256) void gemm_bf_kernel(const __bf16* __restrict__ A,
                                                      const __bf16* __restrict__ B,
                                                      void* __restrict__ OUT) {
  __shared__ __bf16 btile[2][64][KSTAGE];   // 16 KB
  const int tid  = threadIdx.x;
  const int lane = tid & 31;
  const int wave = tid >> 5;
  const int l16  = lane & 15;
  const int hl   = lane >> 4;
  const int n0   = blockIdx.x * 64;
  const int m0   = (blockIdx.y * 8 + wave) * 32;

  // Cooperative async-staging map: 512 x 16B chunks, 2 per thread.
  const int brow = tid >> 2;              // 0..63  (local n)
  const int bcol = (tid & 3) * 16;        // bf16 elements within stage
  const __bf16* bsrc = B + (size_t)(n0 + brow) * D_MODEL + bcol;
  const unsigned lds_base =
      (unsigned)(uintptr_t)&btile[0][brow][bcol];      // LDS offset = addr[31:0]
  const unsigned lds_buf_stride = (unsigned)sizeof(btile[0]);

  v8f c0[4] = {}, c1[4] = {};
  const __bf16* arow0 = A + (size_t)(m0 + l16) * D_MODEL;
  const __bf16* arow1 = arow0 + (size_t)16 * D_MODEL;

  // Prologue: stage 0 into buffer 0.
  async_b128(lds_base, bsrc);
  async_b128(lds_base + 16u, bsrc + 8);

  int buf = 0;
  for (int ks = 0; ks < D_MODEL; ks += KSTAGE, buf ^= 1) {
    if (ks + KSTAGE < D_MODEL) {
      const unsigned l = lds_base + (unsigned)(buf ^ 1) * lds_buf_stride;
      async_b128(l, bsrc + ks + KSTAGE);
      async_b128(l + 16u, bsrc + ks + KSTAGE + 8);
      asm volatile("s_wait_asynccnt 0x2" ::: "memory");  // older pair landed
    } else {
      asm volatile("s_wait_asynccnt 0x0" ::: "memory");
    }
    __syncthreads();   // all waves' chunks of this buffer are in LDS

#pragma unroll
    for (int kc2 = 0; kc2 < KSTAGE; kc2 += 32) {
      __builtin_prefetch(arow0 + ks + kc2 + 256, 0, 1);  // global_prefetch
      const v16bf a0 = ldfrag(arow0 + ks + kc2 + hl * 8,
                              arow0 + ks + kc2 + 16 + hl * 8);
      const v16bf a1 = ldfrag(arow1 + ks + kc2 + hl * 8,
                              arow1 + ks + kc2 + 16 + hl * 8);
#pragma unroll
      for (int t = 0; t < 4; ++t) {
        const __bf16* bp = &btile[buf][t * 16 + l16][kc2 + hl * 8];
        const v16bf b = ldfrag(bp, bp + 16);
        c0[t] = wmma_bf16(a0, b, c0[t]);
        c1[t] = wmma_bf16(a1, b, c1[t]);
      }
    }
    __syncthreads();   // done reading; buffer may be overwritten next step
  }

#pragma unroll
  for (int mt = 0; mt < 2; ++mt) {
#pragma unroll
    for (int r = 0; r < 8; ++r) {
      const int m = m0 + mt * 16 + r + 8 * hl;
#pragma unroll
      for (int t = 0; t < 4; ++t) {
        const float val = (mt == 0) ? c0[t][r] : c1[t][r];
        if (MODE == 0) {
          ((float*)OUT)[(size_t)m * D_MODEL + n0 + t * 16 + l16] = val;
        } else if (MODE == 1) {
          ((__bf16*)OUT)[(size_t)m * D_MODEL + n0 + t * 16 + l16] = f2bf(val);
        } else {  // V^T per head: h = n0/64 (tile stays inside one head)
          ((__bf16*)OUT)[(size_t)(n0 >> 6) * DK * SEQ +
                         (size_t)(t * 16 + l16) * SEQ + m] = f2bf(val);
        }
      }
    }
  }
}

// ---- RoPE in place on bf16 Q/K ([s][h*64+d] layout) ---------------------
__global__ __launch_bounds__(256) void rope_kernel(__bf16* __restrict__ q,
                                                   __bf16* __restrict__ k,
                                                   const int* __restrict__ pos) {
  const int idx = blockIdx.x * blockDim.x + threadIdx.x;
  if (idx >= SEQ * NHEAD * (DK / 2)) return;
  const int j = idx & 31;
  const int h = (idx >> 5) & 15;
  const int s = idx >> 9;
  const float inv = __expf(-((float)(2 * j) / (float)DK) * 9.210340371976184f);
  const float ang = (float)pos[s] * inv;
  float sn, cs;
  __sincosf(ang, &sn, &cs);
  const size_t pidx = ((size_t)s * D_MODEL + h * DK + 2 * j) >> 1;  // dword index
  unsigned* qw = (unsigned*)q;
  unsigned* kw = (unsigned*)k;
  {
    const unsigned w = qw[pidx];
    const float a = __uint_as_float(w << 16);
    const float b = __uint_as_float(w & 0xFFFF0000u);
    qw[pidx] = pack_bf16x2(a * cs - b * sn, b * cs + a * sn);
  }
  {
    const unsigned w = kw[pidx];
    const float a = __uint_as_float(w << 16);
    const float b = __uint_as_float(w & 0xFFFF0000u);
    kw[pidx] = pack_bf16x2(a * cs - b * sn, b * cs + a * sn);
  }
}

// ---- Flash attention, causal. One wave = 16 queries of one head. --------
// Q/K bf16 [s][e]; V bf16 transposed per head [h][dv][seq]; out bf16 [s][e].
__global__ __launch_bounds__(256) void attn_kernel(const __bf16* __restrict__ Q,
                                                   const __bf16* __restrict__ K,
                                                   const __bf16* __restrict__ VT,
                                                   __bf16* __restrict__ O) {
  __shared__ __bf16 plds[8][16][32];
  const int lane = threadIdx.x & 31;
  const int wave = threadIdx.x >> 5;
  const int l16  = lane & 15;
  const int hl   = lane >> 4;
  const int wg   = blockIdx.x * 8 + wave;
  const int h    = wg >> 7;
  const int q0   = (wg & 127) * 16;
  const int hb   = h * DK;

  const __bf16* qrow = Q + (size_t)(q0 + l16) * D_MODEL + hb;
  const v16bf qa0 = ldfrag(qrow + hl * 8, qrow + 16 + hl * 8);
  const v16bf qa1 = ldfrag(qrow + 32 + hl * 8, qrow + 48 + hl * 8);

  v8f o0 = {}, o1 = {}, o2 = {}, o3 = {};
  float mi[8], li[8];
#pragma unroll
  for (int r = 0; r < 8; ++r) { mi[r] = -1e30f; li[r] = 0.0f; }

  const float sscale = 0.125f;  // 1/sqrt(DK)
  const __bf16* vbase = VT + (size_t)h * DK * SEQ + (size_t)l16 * SEQ + hl * 8;

  for (int kb = 0; kb < q0 + 16; kb += 32) {
    v8f s0 = {}, s1 = {};
    {
      const __bf16* k0 = K + (size_t)(kb + l16) * D_MODEL + hb + hl * 8;
      const __bf16* k1 = K + (size_t)(kb + 16 + l16) * D_MODEL + hb + hl * 8;
      s0 = wmma_bf16(qa0, ldfrag(k0, k0 + 16), s0);
      s0 = wmma_bf16(qa1, ldfrag(k0 + 32, k0 + 48), s0);
      s1 = wmma_bf16(qa0, ldfrag(k1, k1 + 16), s1);
      s1 = wmma_bf16(qa1, ldfrag(k1 + 32, k1 + 48), s1);
    }
    float osc[8];
#pragma unroll
    for (int r = 0; r < 8; ++r) {
      const int qi = q0 + r + 8 * hl;
      float x0 = (kb + l16 > qi)      ? -1e30f : s0[r] * sscale;
      float x1 = (kb + 16 + l16 > qi) ? -1e30f : s1[r] * sscale;
      float rm = fmaxf(x0, x1);
#pragma unroll
      for (int m = 1; m < 16; m <<= 1) rm = fmaxf(rm, __shfl_xor(rm, m, 32));
      const float mn = fmaxf(mi[r], rm);
      const float sc = __expf(mi[r] - mn);
      const float p0 = __expf(x0 - mn);
      const float p1 = __expf(x1 - mn);
      float rs = p0 + p1;
#pragma unroll
      for (int m = 1; m < 16; m <<= 1) rs += __shfl_xor(rs, m, 32);
      li[r] = li[r] * sc + rs;
      mi[r] = mn;
      osc[r] = sc;
      plds[wave][r + 8 * hl][l16]      = f2bf(p0);
      plds[wave][r + 8 * hl][16 + l16] = f2bf(p1);
    }
#pragma unroll
    for (int r = 0; r < 8; ++r) {
      o0[r] *= osc[r]; o1[r] *= osc[r]; o2[r] *= osc[r]; o3[r] *= osc[r];
    }
    // Wave-private LDS RAW; LDS is in-order per wave. Enforce + fence compiler.
    asm volatile("s_wait_dscnt 0x0" ::: "memory");
    const v16bf pa = ldfrag(&plds[wave][l16][hl * 8], &plds[wave][l16][16 + hl * 8]);
    const __bf16* vb = vbase + kb;
    o0 = wmma_bf16(pa, ldfrag(vb + 0 * 16 * SEQ, vb + 0 * 16 * SEQ + 16), o0);
    o1 = wmma_bf16(pa, ldfrag(vb + 1 * 16 * SEQ, vb + 1 * 16 * SEQ + 16), o1);
    o2 = wmma_bf16(pa, ldfrag(vb + 2 * 16 * SEQ, vb + 2 * 16 * SEQ + 16), o2);
    o3 = wmma_bf16(pa, ldfrag(vb + 3 * 16 * SEQ, vb + 3 * 16 * SEQ + 16), o3);
    asm volatile("" ::: "memory");
  }

#pragma unroll
  for (int r = 0; r < 8; ++r) {
    const float inv = 1.0f / li[r];
    __bf16* orow = O + (size_t)(q0 + r + 8 * hl) * D_MODEL + hb + l16;
    orow[0]  = f2bf(o0[r] * inv);
    orow[16] = f2bf(o1[r] * inv);
    orow[32] = f2bf(o2[r] * inv);
    orow[48] = f2bf(o3[r] * inv);
  }
}

extern "C" void kernel_launch(void* const* d_in, const int* in_sizes, int n_in,
                              void* d_out, int out_size, void* d_ws, size_t ws_size,
                              hipStream_t stream) {
  const float* x   = (const float*)d_in[0];
  const int*   pos = (const int*)d_in[1];
  const float* WQ  = (const float*)d_in[2];
  const float* WK  = (const float*)d_in[3];
  const float* WV  = (const float*)d_in[4];
  const float* WO  = (const float*)d_in[5];
  float* out = (float*)d_out;

  const size_t MATX = (size_t)SEQ * D_MODEL;      // 2M elems
  const size_t MATW = (size_t)D_MODEL * D_MODEL;  // 1M elems
  __bf16* xbf  = (__bf16*)d_ws;
  __bf16* wqbf = xbf + MATX;
  __bf16* wkbf = wqbf + MATW;
  __bf16* wvbf = wkbf + MATW;
  __bf16* wobf = wvbf + MATW;
  __bf16* qbf  = wobf + MATW;
  __bf16* kbf  = qbf + MATX;
  __bf16* vtbf = kbf + MATX;   // [h][dv][seq]
  __bf16* abf  = vtbf + MATX;

  // One-time operand conversion to bf16.
  cvt_bf16_kernel<<<(int)(MATX / 2 / 256), 256, 0, stream>>>(x, xbf, (int)(MATX / 2));
  cvt_bf16_kernel<<<(int)(MATW / 2 / 256), 256, 0, stream>>>(WQ, wqbf, (int)(MATW / 2));
  cvt_bf16_kernel<<<(int)(MATW / 2 / 256), 256, 0, stream>>>(WK, wkbf, (int)(MATW / 2));
  cvt_bf16_kernel<<<(int)(MATW / 2 / 256), 256, 0, stream>>>(WV, wvbf, (int)(MATW / 2));
  cvt_bf16_kernel<<<(int)(MATW / 2 / 256), 256, 0, stream>>>(WO, wobf, (int)(MATW / 2));

  const dim3 gGrid(D_MODEL / 64, SEQ / 256);  // (16, 8), 256 thr = 8 waves
  gemm_bf_kernel<1><<<gGrid, 256, 0, stream>>>(xbf, wqbf, qbf);
  gemm_bf_kernel<1><<<gGrid, 256, 0, stream>>>(xbf, wkbf, kbf);
  gemm_bf_kernel<2><<<gGrid, 256, 0, stream>>>(xbf, wvbf, vtbf);

  const int ropeN = SEQ * NHEAD * (DK / 2);
  rope_kernel<<<(ropeN + 255) / 256, 256, 0, stream>>>(qbf, kbf, pos);

  attn_kernel<<<(SEQ / 16) * NHEAD / 8, 256, 0, stream>>>(qbf, kbf, vtbf, abf);

  gemm_bf_kernel<0><<<gGrid, 256, 0, stream>>>(abf, wobf, out);
}
